// SelfAttention_69518340653190
// MI455X (gfx1250) — matrix operand order repaired and checked by reading, not compiled, not run on hardware
//
#include <hip/hip_runtime.h>
#include <hip/hip_bf16.h>

typedef _Float16 h16;
typedef __attribute__((ext_vector_type(16))) _Float16 v16h;
typedef __attribute__((ext_vector_type(8)))  _Float16 v8h;
typedef __attribute__((ext_vector_type(8)))  float    v8f;

#define WMMA_F16(a, b, c) \
  __builtin_amdgcn_wmma_f32_16x16x32_f16(false, (a), false, (b), (short)0, (c), false, false)

// ---------------------------------------------------------------------------
// Fragment loaders. All operands are stored K-contiguous ("B stored as B^T").
// A fragment (16x32, MxK): lane L -> row L%16; elements 0..7 = K[8h..8h+7],
// elements 8..15 = K[16+8h..16+8h+7], h = L/16.  (ISA 7.12.2 16-bit A table)
// B fragment (32x16, KxN) loaded from B^T row N: same K order as A.
// C/D (16x16 f32): lane L -> col L%16, VGPR r -> row r + 8*(L/16).
// ---------------------------------------------------------------------------
__device__ __forceinline__ v16h make_v16(v8h lo, v8h hi) {
  v16h r;
#pragma unroll
  for (int i = 0; i < 8; ++i) { r[i] = lo[i]; r[i + 8] = hi[i]; }
  return r;
}
__device__ __forceinline__ v16h frag_a(const h16* row, int lh) {
  v8h lo = *(const v8h*)(row + 8 * lh);
  v8h hi = *(const v8h*)(row + 16 + 8 * lh);
  return make_v16(lo, hi);
}
__device__ __forceinline__ v16h frag_b(const h16* rowT, int lh) {
  v8h lo = *(const v8h*)(rowT + 8 * lh);
  v8h hi = *(const v8h*)(rowT + 16 + 8 * lh);
  return make_v16(lo, hi);
}

// gfx1250 async global->LDS copy, 16 bytes per lane, tracked by ASYNCcnt.
// LDS byte address = low 32 bits of the generic shared-memory pointer
// (flat LDS aperture).
__device__ __forceinline__ void async_g2l_b128(const h16* g, const h16* l) {
  uint32_t loff = (uint32_t)(uintptr_t)l;
  asm volatile("global_load_async_to_lds_b128 %0, %1, off"
               :: "v"(loff), "v"(g)
               : "memory");
}

#define B_    16
#define C_    256
#define N_    1024
#define NH_   4
#define D_    64

// ---------------------------------------------------------------------------
// K0: convert weights f32 -> f16 (qkv_w 768x256, proj_w 256x256; K contiguous)
// ---------------------------------------------------------------------------
__global__ void k_cvt_w(const float* __restrict__ qkv_w,
                        const float* __restrict__ proj_w,
                        h16* __restrict__ qw16, h16* __restrict__ pw16) {
  int i = blockIdx.x * blockDim.x + threadIdx.x;
  int stride = gridDim.x * blockDim.x;
  for (int t = i; t < 3 * C_ * C_; t += stride) qw16[t] = (h16)qkv_w[t];
  for (int t = i; t < C_ * C_; t += stride)     pw16[t] = (h16)proj_w[t];
}

// ---------------------------------------------------------------------------
// K1: per-batch mean / rstd over C*H*W = 262144 elements
// ---------------------------------------------------------------------------
__global__ void k_stats(const float* __restrict__ x, float* __restrict__ stats) {
  __shared__ float sh_s[256], sh_q[256];
  int b = blockIdx.x, tid = threadIdx.x;
  const float* xb = x + (size_t)b * C_ * N_;
  float s = 0.f, q = 0.f;
  for (int i = tid; i < C_ * N_; i += 256) { float v = xb[i]; s += v; q += v * v; }
  sh_s[tid] = s; sh_q[tid] = q; __syncthreads();
  for (int o = 128; o > 0; o >>= 1) {
    if (tid < o) { sh_s[tid] += sh_s[tid + o]; sh_q[tid] += sh_q[tid + o]; }
    __syncthreads();
  }
  if (tid == 0) {
    float inv_n = 1.0f / (float)(C_ * N_);
    float mean = sh_s[0] * inv_n;
    float var  = sh_q[0] * inv_n - mean * mean;
    stats[2 * b]     = mean;
    stats[2 * b + 1] = rsqrtf(var + 1e-5f);
  }
}

// ---------------------------------------------------------------------------
// K2: normalize + affine + transpose (B,C,N) -> xn16 (B,N,C) f16
// ---------------------------------------------------------------------------
__global__ void k_norm_t(const float* __restrict__ x,
                         const float* __restrict__ w, const float* __restrict__ bb,
                         const float* __restrict__ stats, h16* __restrict__ xn16) {
  __shared__ float tile[32][33];
  int b  = blockIdx.z;
  int n0 = blockIdx.x * 32, c0 = blockIdx.y * 32;
  int tx = threadIdx.x, ty = threadIdx.y;
  float mean = stats[2 * b], rstd = stats[2 * b + 1];
#pragma unroll
  for (int k = 0; k < 4; ++k) {
    int c = c0 + ty + k * 8;
    tile[ty + k * 8][tx] = x[((size_t)b * C_ + c) * N_ + n0 + tx];
  }
  __syncthreads();
#pragma unroll
  for (int k = 0; k < 4; ++k) {
    int n = n0 + ty + k * 8;
    int c = c0 + tx;
    float v = (tile[tx][ty + k * 8] - mean) * rstd * w[c] + bb[c];
    xn16[((size_t)b * N_ + n) * C_ + c] = (h16)v;
  }
}

// ---------------------------------------------------------------------------
// Shared GEMM core: 64(M) x 64(N) block tile, 4 waves, K in steps of 32,
// double-buffered LDS fed by async global->LDS copies (ASYNCcnt).
// Each thread issues 4 x b128 async copies per K-step.
// ---------------------------------------------------------------------------
__device__ __forceinline__ void gemm_tile_64x64(
    const h16* __restrict__ A,   // M x 256, K contiguous (row stride 256)
    const h16* __restrict__ BT,  // N x 256, K contiguous (row stride 256)
    int m0, int j0, int tid,
    h16 (*sA)[64 * 32], h16 (*sB)[64 * 32], v8f acc[4]) {
  int lane = tid & 31, wave = tid >> 5;
  int ln = lane & 15, lh = lane >> 4;
  int r  = tid >> 2;         // 0..31
  int cg = (tid & 3) * 8;    // 0,8,16,24 halves

  // prologue: issue K-tile 0 into buffer 0
  {
    async_g2l_b128(&A[(size_t)(m0 + r) * C_ + cg],       &sA[0][r * 32 + cg]);
    async_g2l_b128(&A[(size_t)(m0 + r + 32) * C_ + cg],  &sA[0][(r + 32) * 32 + cg]);
    async_g2l_b128(&BT[(size_t)(j0 + r) * C_ + cg],      &sB[0][r * 32 + cg]);
    async_g2l_b128(&BT[(size_t)(j0 + r + 32) * C_ + cg], &sB[0][(r + 32) * 32 + cg]);
  }
  for (int kt = 0; kt < C_ / 32; ++kt) {
    int buf = kt & 1;
    if (kt + 1 < C_ / 32) {
      int k0 = (kt + 1) * 32;
      int nb = buf ^ 1;
      async_g2l_b128(&A[(size_t)(m0 + r) * C_ + k0 + cg],       &sA[nb][r * 32 + cg]);
      async_g2l_b128(&A[(size_t)(m0 + r + 32) * C_ + k0 + cg],  &sA[nb][(r + 32) * 32 + cg]);
      async_g2l_b128(&BT[(size_t)(j0 + r) * C_ + k0 + cg],      &sB[nb][r * 32 + cg]);
      async_g2l_b128(&BT[(size_t)(j0 + r + 32) * C_ + k0 + cg], &sB[nb][(r + 32) * 32 + cg]);
      asm volatile("s_wait_asynccnt 0x4" ::: "memory");  // tile kt's 4 ops done
    } else {
      asm volatile("s_wait_asynccnt 0x0" ::: "memory");
    }
    __syncthreads();   // all waves' copies into buf are complete
    v16h af = frag_a(&sA[buf][(wave * 16 + ln) * 32], lh);
#pragma unroll
    for (int nt = 0; nt < 4; ++nt) {
      v16h bf = frag_b(&sB[buf][(nt * 16 + ln) * 32], lh);
      acc[nt] = WMMA_F16(af, bf, acc[nt]);
    }
    __syncthreads();   // everyone done reading buf before it is refilled
  }
}

// ---------------------------------------------------------------------------
// K3: QKV GEMM  (1024 x 256) @ (256 x 768) per batch, f16 WMMA, f32 acc.
// Epilogue adds bias, folds 1/sqrt(d) into q, writes v transposed (d,N).
// ---------------------------------------------------------------------------
__global__ void k_qkv(const h16* __restrict__ xn16, const h16* __restrict__ qw16,
                      const float* __restrict__ qkv_b,
                      h16* __restrict__ q, h16* __restrict__ kk, h16* __restrict__ vT) {
  __shared__ h16 sA[2][64 * 32];
  __shared__ h16 sB[2][64 * 32];
  int b  = blockIdx.z;
  int m0 = blockIdx.y * 64;
  int j0 = blockIdx.x * 64;
  int tid = threadIdx.x, lane = tid & 31, wave = tid >> 5;
  int ln = lane & 15, lh = lane >> 4;
  v8f zero8 = {0.f, 0.f, 0.f, 0.f, 0.f, 0.f, 0.f, 0.f};
  v8f acc[4] = {zero8, zero8, zero8, zero8};

  gemm_tile_64x64(xn16 + (size_t)b * N_ * C_, qw16, m0, j0, tid, sA, sB, acc);

#pragma unroll
  for (int nt = 0; nt < 4; ++nt) {
#pragma unroll
    for (int rr = 0; rr < 8; ++rr) {
      int m = m0 + wave * 16 + rr + 8 * lh;   // token index
      int j = j0 + nt * 16 + ln;              // qkv output col
      float v = acc[nt][rr] + qkv_b[j];
      int head = j / (3 * D_), rem = j % (3 * D_);
      if (rem < D_) {
        q[(((size_t)b * NH_ + head) * N_ + m) * D_ + rem] = (h16)(v * 0.125f);
      } else if (rem < 2 * D_) {
        kk[(((size_t)b * NH_ + head) * N_ + m) * D_ + (rem - D_)] = (h16)v;
      } else {
        vT[(((size_t)b * NH_ + head) * D_ + (rem - 2 * D_)) * N_ + m] = (h16)v;
      }
    }
  }
}

// ---------------------------------------------------------------------------
// K4: flash-style attention per (b, head). Block = 4 waves, 64 query rows
// (16 per wave). Online softmax; P goes through LDS to re-layout D->A frag.
// ---------------------------------------------------------------------------
__global__ void k_attn(const h16* __restrict__ q, const h16* __restrict__ kk,
                       const h16* __restrict__ vT, h16* __restrict__ o16) {
  __shared__ h16 sP[64 * 64];
  int b = blockIdx.z, head = blockIdx.y, m0 = blockIdx.x * 64;
  int tid = threadIdx.x, lane = tid & 31, wave = tid >> 5;
  int ln = lane & 15, lh = lane >> 4;
  const h16* qh = q  + ((size_t)b * NH_ + head) * N_ * D_;
  const h16* kh = kk + ((size_t)b * NH_ + head) * N_ * D_;
  const h16* vh = vT + ((size_t)b * NH_ + head) * D_ * N_;

  const h16* qrow = qh + (size_t)(m0 + wave * 16 + ln) * D_;
  v16h qf0 = frag_a(qrow, lh);        // K = 0..31 of d
  v16h qf1 = frag_a(qrow + 32, lh);   // K = 32..63 of d

  v8f zero8 = {0.f, 0.f, 0.f, 0.f, 0.f, 0.f, 0.f, 0.f};
  v8f accO[4] = {zero8, zero8, zero8, zero8};
  float mi[8], li[8];
#pragma unroll
  for (int rr = 0; rr < 8; ++rr) { mi[rr] = -1e30f; li[rr] = 0.f; }

  for (int c0 = 0; c0 < N_; c0 += 64) {
    // --- S = q @ k^T (scaled q), 16x64 per wave ---
    v8f s[4] = {zero8, zero8, zero8, zero8};
#pragma unroll
    for (int nt = 0; nt < 4; ++nt) {
      const h16* krow = kh + (size_t)(c0 + nt * 16 + ln) * D_;
      v16h b0 = frag_b(krow, lh);
      v16h b1 = frag_b(krow + 32, lh);
      s[nt] = WMMA_F16(qf0, b0, s[nt]);
      s[nt] = WMMA_F16(qf1, b1, s[nt]);
    }
    // --- online softmax update (row = rr + 8*lh; 16 lanes per row-half) ---
    float alpha[8];
#pragma unroll
    for (int rr = 0; rr < 8; ++rr) {
      float mx = fmaxf(fmaxf(s[0][rr], s[1][rr]), fmaxf(s[2][rr], s[3][rr]));
#pragma unroll
      for (int off = 1; off < 16; off <<= 1) mx = fmaxf(mx, __shfl_xor(mx, off, 32));
      float mnew = fmaxf(mi[rr], mx);
      alpha[rr] = __expf(mi[rr] - mnew);
      float rs = 0.f;
#pragma unroll
      for (int nt = 0; nt < 4; ++nt) {
        float p = __expf(s[nt][rr] - mnew);
        s[nt][rr] = p;
        rs += p;
      }
#pragma unroll
      for (int off = 1; off < 16; off <<= 1) rs += __shfl_xor(rs, off, 32);
      li[rr] = li[rr] * alpha[rr] + rs;
      mi[rr] = mnew;
    }
    // --- P (f16) -> LDS in row-major so it can be re-read as A fragments ---
#pragma unroll
    for (int nt = 0; nt < 4; ++nt)
#pragma unroll
      for (int rr = 0; rr < 8; ++rr)
        sP[(wave * 16 + rr + 8 * lh) * 64 + nt * 16 + ln] = (h16)s[nt][rr];
    __syncthreads();
    // --- O = diag(alpha)*O + P @ V ---
    const h16* prow = &sP[(wave * 16 + ln) * 64];
    v16h pf0 = frag_a(prow, lh);
    v16h pf1 = frag_a(prow + 32, lh);
#pragma unroll
    for (int dt = 0; dt < 4; ++dt) {
#pragma unroll
      for (int rr = 0; rr < 8; ++rr) accO[dt][rr] *= alpha[rr];
      const h16* vrow = vh + (size_t)(dt * 16 + ln) * N_ + c0;
      v16h vb0 = frag_b(vrow, lh);
      v16h vb1 = frag_b(vrow + 32, lh);
      accO[dt] = WMMA_F16(pf0, vb0, accO[dt]);
      accO[dt] = WMMA_F16(pf1, vb1, accO[dt]);
    }
    __syncthreads();
  }
  // --- normalize and store to o16 (B,N,C) with C index = head*64 + d ---
#pragma unroll
  for (int dt = 0; dt < 4; ++dt)
#pragma unroll
    for (int rr = 0; rr < 8; ++rr) {
      int n = m0 + wave * 16 + rr + 8 * lh;
      int c = head * D_ + dt * 16 + ln;
      o16[((size_t)b * N_ + n) * C_ + c] = (h16)(accO[dt][rr] / li[rr]);
    }
}

// ---------------------------------------------------------------------------
// K5: out-projection GEMM + bias + residual, write f32 (B,C,H,W)
// ---------------------------------------------------------------------------
__global__ void k_proj(const h16* __restrict__ o16, const h16* __restrict__ pw16,
                       const float* __restrict__ proj_b, const float* __restrict__ x,
                       float* __restrict__ out) {
  __shared__ h16 sA[2][64 * 32];
  __shared__ h16 sB[2][64 * 32];
  int b  = blockIdx.z;
  int m0 = blockIdx.y * 64;
  int j0 = blockIdx.x * 64;
  int tid = threadIdx.x, lane = tid & 31, wave = tid >> 5;
  int ln = lane & 15, lh = lane >> 4;
  v8f zero8 = {0.f, 0.f, 0.f, 0.f, 0.f, 0.f, 0.f, 0.f};
  v8f acc[4] = {zero8, zero8, zero8, zero8};

  gemm_tile_64x64(o16 + (size_t)b * N_ * C_, pw16, m0, j0, tid, sA, sB, acc);

#pragma unroll
  for (int nt = 0; nt < 4; ++nt) {
#pragma unroll
    for (int rr = 0; rr < 8; ++rr) {
      int m = m0 + wave * 16 + rr + 8 * lh;   // token index n
      int j = j0 + nt * 16 + ln;              // channel c
      size_t xi = ((size_t)b * C_ + j) * N_ + m;
      out[xi] = x[xi] + acc[nt][rr] + proj_b[j];
    }
  }
}

// ---------------------------------------------------------------------------
extern "C" void kernel_launch(void* const* d_in, const int* in_sizes, int n_in,
                              void* d_out, int out_size, void* d_ws, size_t ws_size,
                              hipStream_t stream) {
  const float* x      = (const float*)d_in[0];
  const float* norm_w = (const float*)d_in[1];
  const float* norm_b = (const float*)d_in[2];
  const float* qkv_w  = (const float*)d_in[3];
  const float* qkv_b  = (const float*)d_in[4];
  const float* proj_w = (const float*)d_in[5];
  const float* proj_b = (const float*)d_in[6];
  float* out = (float*)d_out;
  char*  ws  = (char*)d_ws;

  h16*   qw16  = (h16*)  (ws + 0x0000000);  // 768*256 f16
  h16*   pw16  = (h16*)  (ws + 0x0080000);  // 256*256 f16
  float* stats = (float*)(ws + 0x00A0000);  // 16 * {mean, rstd}
  h16*   xn16  = (h16*)  (ws + 0x0100000);  // (B,N,C) 8 MiB
  h16*   qbuf  = (h16*)  (ws + 0x0900000);  // (B,h,N,d) 8 MiB
  h16*   kbuf  = (h16*)  (ws + 0x1100000);  // (B,h,N,d) 8 MiB
  h16*   vTbuf = (h16*)  (ws + 0x1900000);  // (B,h,d,N) 8 MiB
  h16*   obuf  = (h16*)  (ws + 0x2100000);  // (B,N,C) 8 MiB

  k_cvt_w<<<256, 256, 0, stream>>>(qkv_w, proj_w, qw16, pw16);
  k_stats<<<B_, 256, 0, stream>>>(x, stats);
  k_norm_t<<<dim3(N_ / 32, C_ / 32, B_), dim3(32, 8), 0, stream>>>(
      x, norm_w, norm_b, stats, xn16);
  k_qkv<<<dim3(3 * C_ / 64, N_ / 64, B_), 128, 0, stream>>>(
      xn16, qw16, qkv_b, qbuf, kbuf, vTbuf);
  k_attn<<<dim3(N_ / 64, NH_, B_), 128, 0, stream>>>(qbuf, kbuf, vTbuf, obuf);
  k_proj<<<dim3(C_ / 64, N_ / 64, B_), 128, 0, stream>>>(
      obuf, pw16, proj_b, x, out);
}
